// FPModule_19344532701796
// MI455X (gfx1250) — compile-verified, matrix operand order checked by reading, not run on hardware
//
#include <hip/hip_runtime.h>

typedef __attribute__((ext_vector_type(16))) _Float16 v16h;
typedef __attribute__((ext_vector_type(8)))  _Float16 v8h;
typedef __attribute__((ext_vector_type(8)))  float    v8f;

constexpr int B_     = 8;
constexpr int NC     = 2048;
constexpr int NF     = 8192;
constexpr int CIN    = 256;
constexpr int CSK    = 128;
constexpr int COUT   = 256;
constexpr int KH     = CIN + CSK;   // 384
constexpr int KSTEPS = KH / 32;     // 12 WMMA K-steps
constexpr int NTILES = COUT / 16;   // 16 N-tiles
constexpr int MROWS  = 32;          // fine rows per workgroup (2 M-tiles)
constexpr int HS     = 392;         // LDS h-tile row stride in halfs (16B aligned, conflict-free)
constexpr float EPSF = 1e-16f;

// ---------------------------------------------------------------------------
// Kernel 1: 3-NN search. One thread per fine point; per-batch coarse positions
// staged in LDS (24KB). Uniform scan index -> LDS broadcast reads.
// ---------------------------------------------------------------------------
__global__ __launch_bounds__(256) void knn_kernel(
    const float* __restrict__ pos, const float* __restrict__ pos_skip,
    int* __restrict__ oidx, float* __restrict__ ow) {
  __shared__ float lp[NC * 3];
  const int tid = threadIdx.x;
  const int b   = blockIdx.x / (NF / 256);
  const float* pb = pos + (size_t)b * NC * 3;
  for (int i = tid; i < NC * 3; i += 256) lp[i] = pb[i];
  __syncthreads();

  const int g = blockIdx.x * 256 + tid;
  const float qx = pos_skip[3*g+0];
  const float qy = pos_skip[3*g+1];
  const float qz = pos_skip[3*g+2];

  float d0 = 3.4e38f, d1 = 3.4e38f, d2 = 3.4e38f;
  int   i0 = 0, i1 = 0, i2 = 0;
  #pragma unroll 4
  for (int j = 0; j < NC; ++j) {
    float dx = qx - lp[3*j+0];
    float dy = qy - lp[3*j+1];
    float dz = qz - lp[3*j+2];
    float d  = fmaf(dx, dx, fmaf(dy, dy, dz * dz));
    if (d < d0)      { d2 = d1; i2 = i1; d1 = d0; i1 = i0; d0 = d; i0 = j; }
    else if (d < d1) { d2 = d1; i2 = i1; d1 = d;  i1 = j; }
    else if (d < d2) { d2 = d;  i2 = j; }
  }
  float w0 = 1.f / fmaxf(d0, EPSF);
  float w1 = 1.f / fmaxf(d1, EPSF);
  float w2 = 1.f / fmaxf(d2, EPSF);
  float inv = 1.f / (w0 + w1 + w2);
  oidx[4*g+0] = b * NC + i0;
  oidx[4*g+1] = b * NC + i1;
  oidx[4*g+2] = b * NC + i2;
  ow[4*g+0] = w0 * inv;
  ow[4*g+1] = w1 * inv;
  ow[4*g+2] = w2 * inv;
}

// ---------------------------------------------------------------------------
// Kernel 2: swizzle W [384x256] f32 -> f16 in WMMA B-fragment order so that in
// the GEMM each lane's 16 halfs are a single contiguous 32B load (L2-resident).
// Layout: Wp[ks][ntile][lane][i], i -> k = ks*32 + kb + (i&7) + (i>=8?16:0),
// kb = lane<16 ? 0 : 8, n = ntile*16 + (lane&15).
// ---------------------------------------------------------------------------
__global__ void prep_w(const float* __restrict__ W, _Float16* __restrict__ Wp) {
  int t = blockIdx.x * 256 + threadIdx.x;
  if (t >= KSTEPS * NTILES * 32 * 16) return;
  int i    = t & 15;
  int lane = (t >> 4) & 31;
  int nt   = (t >> 9) & 15;
  int ks   = t >> 13;
  int n    = nt * 16 + (lane & 15);
  int kb   = (lane < 16) ? 0 : 8;
  int k    = ks * 32 + kb + (i & 7) + ((i >= 8) ? 16 : 0);
  Wp[t] = (_Float16)W[(size_t)k * COUT + n];
}

// ---------------------------------------------------------------------------
// Kernel 3: fused interpolate + concat + GEMM(WMMA f16->f32) + bias + ReLU.
// Workgroup = 256 threads = 8 waves; 32 fine rows (2 M-tiles) x 256 cols.
// Each wave: 2 M-tiles x 2 N-tiles -> B fragments reused across M (halves L2
// traffic for W), A fragments reused across N. 48 WMMAs per wave.
// ---------------------------------------------------------------------------
__global__ __launch_bounds__(256) void fused_kernel(
    const float* __restrict__ x, const float* __restrict__ x_skip,
    const int* __restrict__ kidx, const float* __restrict__ kw,
    const v16h* __restrict__ Wp, const float* __restrict__ bias,
    float* __restrict__ out) {
  __shared__ _Float16 h[MROWS * HS];

  const int tid      = threadIdx.x;
  const int tileBase = blockIdx.x * MROWS;

  // --- build h tile (two 16-row passes; 16 threads per row, 64B coalesced) ---
  #pragma unroll
  for (int rb = 0; rb < MROWS; rb += 16) {
    const int r  = rb + (tid >> 4);
    const int c0 = tid & 15;
    const int g  = tileBase + r;
    const int ia = kidx[4*g+0], ib = kidx[4*g+1], ic = kidx[4*g+2];
    const float wa = kw[4*g+0], wb = kw[4*g+1], wc = kw[4*g+2];
    const float* xa = x + (size_t)ia * CIN;
    const float* xb = x + (size_t)ib * CIN;
    const float* xc = x + (size_t)ic * CIN;
    for (int c = c0; c < CIN; c += 16) {
      float v = wa * xa[c] + wb * xb[c] + wc * xc[c];
      h[r * HS + c] = (_Float16)v;
    }
    const float* xs = x_skip + (size_t)g * CSK;
    for (int c = c0; c < CSK; c += 16) {
      h[r * HS + CIN + c] = (_Float16)xs[c];
    }
  }
  __syncthreads();

  // --- WMMA: 2 M-tiles x 2 N-tiles per wave over K=384 ---
  const int lane = tid & 31;
  const int wave = tid >> 5;
  const int nt0  = wave * 2;
  const int nt1  = wave * 2 + 1;
  const int kb   = (lane < 16) ? 0 : 8;
  const _Float16* hrow0 = &h[(lane & 15) * HS];        // M-tile 0: rows 0..15
  const _Float16* hrow1 = hrow0 + 16 * HS;             // M-tile 1: rows 16..31

  v8f acc00 = {}, acc01 = {}, acc10 = {}, acc11 = {};

  #pragma unroll
  for (int ks = 0; ks < KSTEPS; ++ks) {
    union { v16h v; v8h p[2]; } a0, a1;
    a0.p[0] = *(const v8h*)(hrow0 + ks * 32 + kb);       // ds_read_b128
    a0.p[1] = *(const v8h*)(hrow0 + ks * 32 + kb + 16);  // ds_read_b128
    a1.p[0] = *(const v8h*)(hrow1 + ks * 32 + kb);
    a1.p[1] = *(const v8h*)(hrow1 + ks * 32 + kb + 16);
    v16h b0 = Wp[(ks * NTILES + nt0) * 32 + lane];       // 32B coalesced, L2-hot
    v16h b1 = Wp[(ks * NTILES + nt1) * 32 + lane];
    acc00 = __builtin_amdgcn_wmma_f32_16x16x32_f16(
        false, a0.v, false, b0, (short)0, acc00, false, false);
    acc01 = __builtin_amdgcn_wmma_f32_16x16x32_f16(
        false, a0.v, false, b1, (short)0, acc01, false, false);
    acc10 = __builtin_amdgcn_wmma_f32_16x16x32_f16(
        false, a1.v, false, b0, (short)0, acc10, false, false);
    acc11 = __builtin_amdgcn_wmma_f32_16x16x32_f16(
        false, a1.v, false, b1, (short)0, acc11, false, false);
  }

  // --- bias + ReLU + store (C/D layout: VGPR r -> M = r + 8*(lane>=16)) ---
  const int n0  = nt0 * 16 + (lane & 15);
  const int n1  = nt1 * 16 + (lane & 15);
  const float bz0 = bias[n0];
  const float bz1 = bias[n1];
  const int mh = (lane >> 4) << 3;
  #pragma unroll
  for (int rr = 0; rr < 8; ++rr) {
    const int g0 = tileBase + rr + mh;        // M-tile 0 row
    const int g1 = g0 + 16;                   // M-tile 1 row
    float v00 = acc00[rr] + bz0;
    float v01 = acc01[rr] + bz1;
    float v10 = acc10[rr] + bz0;
    float v11 = acc11[rr] + bz1;
    out[(size_t)g0 * COUT + n0] = v00 > 0.f ? v00 : 0.f;
    out[(size_t)g0 * COUT + n1] = v01 > 0.f ? v01 : 0.f;
    out[(size_t)g1 * COUT + n0] = v10 > 0.f ? v10 : 0.f;
    out[(size_t)g1 * COUT + n1] = v11 > 0.f ? v11 : 0.f;
  }
}

// ---------------------------------------------------------------------------
// Generic 32-bit word copy for the tuple tail (pos_skip, batch_skip pass-through)
// ---------------------------------------------------------------------------
__global__ void copy32(const unsigned* __restrict__ src, unsigned* __restrict__ dst, long n) {
  long i = (long)blockIdx.x * blockDim.x + threadIdx.x;
  if (i < n) dst[i] = src[i];
}

extern "C" void kernel_launch(void* const* d_in, const int* in_sizes, int n_in,
                              void* d_out, int out_size, void* d_ws, size_t ws_size,
                              hipStream_t stream) {
  (void)in_sizes; (void)n_in; (void)ws_size;

  const float* x          = (const float*)d_in[0];
  const float* pos        = (const float*)d_in[1];
  // d_in[2] = batch (implicit via contiguous layout)
  const float* x_skip     = (const float*)d_in[3];
  const float* pos_skip   = (const float*)d_in[4];
  const void*  batch_skip = d_in[5];
  const float* W          = (const float*)d_in[6];
  const float* bias       = (const float*)d_in[7];
  float*       out        = (float*)d_out;

  // workspace layout: [ Wp f16 swizzled | knn idx (int4/pt) | knn w (float4/pt) ]
  char* ws = (char*)d_ws;
  const size_t wpElems = (size_t)KSTEPS * NTILES * 32 * 16;   // 98304 halfs
  _Float16* Wp   = (_Float16*)ws;
  int*      kidx = (int*)(ws + wpElems * sizeof(_Float16));
  float*    kw   = (float*)(ws + wpElems * sizeof(_Float16)
                               + (size_t)B_ * NF * 4 * sizeof(int));

  const int nfTotal = B_ * NF;

  knn_kernel<<<nfTotal / 256, 256, 0, stream>>>(pos, pos_skip, kidx, kw);
  prep_w<<<(int)((wpElems + 255) / 256), 256, 0, stream>>>(W, Wp);
  fused_kernel<<<nfTotal / MROWS, 256, 0, stream>>>(x, x_skip, kidx, kw,
                                                    (const v16h*)Wp, bias, out);

  // tuple tail: pos_skip then batch_skip, flattened after `out`
  const long outElems = (long)nfTotal * COUT;
  const long posElems = (long)nfTotal * 3;
  copy32<<<(int)((posElems + 255) / 256), 256, 0, stream>>>(
      (const unsigned*)pos_skip, (unsigned*)out + outElems, posElems);
  long rem = (long)out_size - outElems - posElems;
  if (rem > 0) {
    long words = rem;
    const long maxw = (long)nfTotal * 2;   // int64 batch ids as 32-bit words
    if (words > maxw) words = maxw;
    copy32<<<(int)((words + 255) / 256), 256, 0, stream>>>(
        (const unsigned*)batch_skip, (unsigned*)out + outElems + posElems, words);
  }
}
